// Transformer_69776038691015
// MI455X (gfx1250) — matrix operand order repaired and checked by reading, not compiled, not run on hardware
//
#include <hip/hip_runtime.h>
#include <hip/hip_bf16.h>
#include <math.h>

// ---------------------------------------------------------------------------
// Transformer forward on gfx1250 (CDNA5), bf16 storage everywhere.
// All GEMMs: v_wmma_f32_16x16x32_bf16, pure-b128 LDS staging (no conversion
// in the hot loop). Weights transposed+converted once per layer per call.
// Shapes: B=1, T=2048, D=512, H=8, hd=64, L=6, DFF=2048, V=100.
// ---------------------------------------------------------------------------

typedef __attribute__((ext_vector_type(16))) __bf16 v16bf;
typedef __attribute__((ext_vector_type(8)))  __bf16 v8bf;
typedef __attribute__((ext_vector_type(8)))  float  v8f;

#define Dd   512
#define NTHR 256

#define MODE_BF16  0   // bf16 row-major out
#define MODE_BF16T 1   // bf16 transposed out (Y[n][m]) -- packed uint4 stores
#define MODE_F32   2   // fp32 row-major out

__device__ __forceinline__ float gelu_f(float x) {
    return 0.5f * x * (1.0f + erff(x * 0.70710678118654752f));
}

__device__ __forceinline__ unsigned pk(float a, float b) {
    union { __bf16 h[2]; unsigned u; } p;
    p.h[0] = (__bf16)a; p.h[1] = (__bf16)b;
    return p.u;
}

__device__ __forceinline__ v16bf mrg(v8bf lo, v8bf hi) {
    return __builtin_shufflevector(lo, hi, 0,1,2,3,4,5,6,7,8,9,10,11,12,13,14,15);
}

// ---------------------------------------------------------------------------
// Transpose + convert: fp32 in[K][N] -> bf16 out[N][K].  Tiled 32x32 via LDS.
// ---------------------------------------------------------------------------
__global__ __launch_bounds__(NTHR) void convT_kernel(
    const float* __restrict__ in, __bf16* __restrict__ out, int K, int N)
{
    __shared__ float tile[32][33];
    int n0 = blockIdx.x * 32, k0 = blockIdx.y * 32;
    int tx = threadIdx.x & 31, ty = threadIdx.x >> 5;
    #pragma unroll
    for (int i = 0; i < 4; i++) {
        int kk = ty * 4 + i;
        tile[kk][tx] = (n0 + tx < N) ? in[(size_t)(k0 + kk) * N + n0 + tx] : 0.0f;
    }
    __syncthreads();
    #pragma unroll
    for (int i = 0; i < 4; i++) {
        int nn = ty * 4 + i;
        if (n0 + nn < N) out[(size_t)(n0 + nn) * K + k0 + tx] = (__bf16)tile[tx][nn];
    }
}

// ---------------------------------------------------------------------------
// Embedding: h[t,:] = bf16(emb[x[t],:] + pos[t,:])
// ---------------------------------------------------------------------------
__global__ __launch_bounds__(NTHR) void embed_kernel(
    const int* __restrict__ x, const float* __restrict__ emb,
    const float* __restrict__ pos, __bf16* __restrict__ h)
{
    int t = blockIdx.x, d = threadIdx.x;
    int tok = x[t];
    size_t ro = (size_t)t * Dd;
    h[ro + d]       = (__bf16)(emb[(size_t)tok * Dd + d]       + pos[ro + d]);
    h[ro + d + 256] = (__bf16)(emb[(size_t)tok * Dd + d + 256] + pos[ro + d + 256]);
}

// ---------------------------------------------------------------------------
// LayerNorm over D=512 per row: bf16 in -> fp32 stats -> bf16 out.
// ---------------------------------------------------------------------------
__global__ __launch_bounds__(NTHR) void ln_kernel(
    const __bf16* __restrict__ X, __bf16* __restrict__ Y,
    const float* __restrict__ g, const float* __restrict__ b)
{
    int row = blockIdx.x, t = threadIdx.x;
    const __bf16* xr = X + (size_t)row * Dd;
    float v0 = (float)xr[t], v1 = (float)xr[t + 256];
    __shared__ float red[NTHR];
    red[t] = v0 + v1;
    __syncthreads();
    for (int s = 128; s > 0; s >>= 1) { if (t < s) red[t] += red[t + s]; __syncthreads(); }
    float mean = red[0] * (1.0f / 512.0f);
    __syncthreads();
    float d0 = v0 - mean, d1 = v1 - mean;
    red[t] = d0 * d0 + d1 * d1;
    __syncthreads();
    for (int s = 128; s > 0; s >>= 1) { if (t < s) red[t] += red[t + s]; __syncthreads(); }
    float rstd = rsqrtf(red[0] * (1.0f / 512.0f) + 1e-5f);
    __bf16* yr = Y + (size_t)row * Dd;
    yr[t]       = (__bf16)(d0 * rstd * g[t]       + b[t]);
    yr[t + 256] = (__bf16)(d1 * rstd * g[t + 256] + b[t + 256]);
}

// ---------------------------------------------------------------------------
// Causal softmax: fp32 scores row i (cols [0,i]) -> bf16 probs, tail zeroed.
// ---------------------------------------------------------------------------
__global__ __launch_bounds__(NTHR) void softmax_row_kernel(
    const float* __restrict__ S, __bf16* __restrict__ P, int ld)
{
    int i = blockIdx.x, t = threadIdx.x;
    const float* row = S + (size_t)i * ld;
    __bf16* prow = P + (size_t)i * ld;
    int n = i + 1;
    __shared__ float red[NTHR];
    float m = -3.4e38f;
    for (int j = t; j < n; j += NTHR) m = fmaxf(m, row[j]);
    red[t] = m;
    __syncthreads();
    for (int s = 128; s > 0; s >>= 1) { if (t < s) red[t] = fmaxf(red[t], red[t + s]); __syncthreads(); }
    m = red[0];
    __syncthreads();
    float sum = 0.0f;
    for (int j = t; j < n; j += NTHR) sum += expf(row[j] - m);
    red[t] = sum;
    __syncthreads();
    for (int s = 128; s > 0; s >>= 1) { if (t < s) red[t] += red[t + s]; __syncthreads(); }
    float inv = 1.0f / red[0];
    for (int j = t; j < n; j += NTHR) prow[j] = (__bf16)(expf(row[j] - m) * inv);
    for (int j = n + t; j < ld; j += NTHR) prow[j] = (__bf16)0.0f;
}

// ---------------------------------------------------------------------------
// WMMA GEMM (all-bf16 operands): Y = act(X[M,K] @ W + bias (+ res))
// W given TRANSPOSED: Wt[N][K] bf16, row stride = K.
// Tile 128x64, 8 waves, wave owns 32x32 (4 wmma / K-chunk of 32).
// Staging is pure b128 copies: global b128 -> LDS b128 (no conversions).
// ---------------------------------------------------------------------------
__global__ __launch_bounds__(NTHR) void gemm_wmma_kernel(
    const __bf16* __restrict__ X, int lda,
    const __bf16* __restrict__ Wt,
    const float* __restrict__ bias,
    void* __restrict__ Yv, int ldc,
    const __bf16* __restrict__ res,
    int K, int Nreal, int mode, int act)
{
    __shared__ __align__(16) __bf16 As[128][32];  // [row][k]  pitch 64B
    __shared__ __align__(16) __bf16 Bt[64][40];   // [col][k]  pitch 80B

    const int tid  = threadIdx.x;
    const int lane = tid & 31, w = tid >> 5;
    const int wr = w & 3, wc = w >> 2;
    const int tile_m = blockIdx.x * 128, tile_n = blockIdx.y * 64;

    v8f acc00 = {}, acc01 = {}, acc10 = {}, acc11 = {};

    const int ar = tid >> 1, acs = (tid & 1) * 16;  // A: row, 16-elem seg
    const int bc = tid >> 2, bks = (tid & 3) * 8;   // B: col, 8-k seg
    const int gcol = tile_n + bc;

    const int l15  = lane & 15;
    const int khiA = (lane >= 16) ? 8 : 0;
    const int kbB  = (lane >= 16) ? 16 : 0;

    const __bf16* arow_g = X + (size_t)(tile_m + ar) * lda + acs;
    const __bf16* brow_g = Wt + (size_t)gcol * K + bks;

    for (int k0 = 0; k0 < K; k0 += 32) {
        // ---- stage A (128x32 bf16): two 16B copies per thread ----
        const uint4* ap = (const uint4*)(arow_g + k0);
        uint4 a0 = ap[0];
        uint4 a1 = ap[1];
        *(uint4*)&As[ar][acs]     = a0;
        *(uint4*)&As[ar][acs + 8] = a1;

        // ---- stage B (64 cols x 32 k): one 16B copy per thread ----
        uint4 bvv = {0u, 0u, 0u, 0u};
        if (gcol < Nreal) bvv = *(const uint4*)(brow_g + k0);
        *(uint4*)&Bt[bc][bks] = bvv;

        if (k0 + 32 < K) {
            __builtin_prefetch((const void*)(arow_g + k0 + 32), 0, 0);
            if (gcol < Nreal) __builtin_prefetch((const void*)(brow_g + k0 + 32), 0, 0);
        }
        __syncthreads();

        // ---- fragments: contiguous 16B LDS reads ----
        v16bf af0, af1, bf0, bf1;
        {
            const __bf16* p = &As[wr * 32 + l15][0];
            af0 = mrg(*(const v8bf*)(p + khiA), *(const v8bf*)(p + 16 + khiA));
            p = &As[wr * 32 + 16 + l15][0];
            af1 = mrg(*(const v8bf*)(p + khiA), *(const v8bf*)(p + 16 + khiA));
            const __bf16* q = &Bt[wc * 32 + l15][0];
            bf0 = mrg(*(const v8bf*)(q + khiA), *(const v8bf*)(q + 16 + khiA));
            q = &Bt[wc * 32 + 16 + l15][0];
            bf1 = mrg(*(const v8bf*)(q + khiA), *(const v8bf*)(q + 16 + khiA));
        }
        acc00 = __builtin_amdgcn_wmma_f32_16x16x32_bf16(false, af0, false, bf0, (short)0, acc00, false, false);
        acc01 = __builtin_amdgcn_wmma_f32_16x16x32_bf16(false, af0, false, bf1, (short)0, acc01, false, false);
        acc10 = __builtin_amdgcn_wmma_f32_16x16x32_bf16(false, af1, false, bf0, (short)0, acc10, false, false);
        acc11 = __builtin_amdgcn_wmma_f32_16x16x32_bf16(false, af1, false, bf1, (short)0, acc11, false, false);
        __syncthreads();
    }

    // ---- epilogue ----
    const int lhi = (lane >= 16) ? 8 : 0;
    __bf16* Yb = (__bf16*)Yv;
    float*  Yf = (float*)Yv;

    #pragma unroll
    for (int ai = 0; ai < 2; ai++) {
        int mb = tile_m + wr * 32 + ai * 16 + lhi;
        #pragma unroll
        for (int bi = 0; bi < 2; bi++) {
            int n = tile_n + wc * 32 + bi * 16 + l15;
            if (n < Nreal) {
                v8f a = ai ? (bi ? acc11 : acc10) : (bi ? acc01 : acc00);
                float bv = bias ? bias[n] : 0.0f;
                float v[8];
                #pragma unroll
                for (int r = 0; r < 8; r++) {
                    float t = a[r] + bv;
                    if (res) t += (float)res[(size_t)(mb + r) * ldc + n];
                    if (act == 1) t = gelu_f(t);
                    v[r] = t;
                }
                if (mode == MODE_BF16T) {
                    // Y[n][mb..mb+7]: 8 consecutive bf16 -> one 16B store
                    uint4 pv;
                    pv.x = pk(v[0], v[1]); pv.y = pk(v[2], v[3]);
                    pv.z = pk(v[4], v[5]); pv.w = pk(v[6], v[7]);
                    *(uint4*)&Yb[(size_t)n * ldc + mb] = pv;
                } else if (mode == MODE_F32) {
                    #pragma unroll
                    for (int r = 0; r < 8; r++)
                        Yf[(size_t)(mb + r) * ldc + n] = v[r];
                } else {
                    #pragma unroll
                    for (int r = 0; r < 8; r++)
                        Yb[(size_t)(mb + r) * ldc + n] = (__bf16)v[r];
                }
            }
        }
    }
}

// ---------------------------------------------------------------------------
// Attention scores: S[i,j] = scale * dot(Q[i,0:64], K[j,0:64]), fp32 out.
// Q,K bf16 row-major (head slice), so BOTH tiles stage as pure b128 copies
// (K rows are exactly the Bt[col][k] layout). Causal-skip above diagonal.
// ---------------------------------------------------------------------------
__global__ __launch_bounds__(NTHR) void attn_score_kernel(
    const __bf16* __restrict__ Q, const __bf16* __restrict__ Km, int ld,
    float* __restrict__ S, int ldS, float scale)
{
    if (blockIdx.y > blockIdx.x) return;

    __shared__ __align__(16) __bf16 As[64][32];
    __shared__ __align__(16) __bf16 Bt[64][40];

    const int tid  = threadIdx.x;
    const int lane = tid & 31, w = tid >> 5;
    const int wr = w & 3, wc = w >> 2;
    const int tile_m = blockIdx.x * 64, tile_n = blockIdx.y * 64;

    v8f acc0 = {}, acc1 = {};

    const int ar = tid >> 2, ks8 = (tid & 3) * 8;

    const int l15  = lane & 15;
    const int khiA = (lane >= 16) ? 8 : 0;
    const int kbB  = (lane >= 16) ? 16 : 0;

    for (int k0 = 0; k0 < 64; k0 += 32) {
        *(uint4*)&As[ar][ks8] = *(const uint4*)(Q  + (size_t)(tile_m + ar) * ld + k0 + ks8);
        *(uint4*)&Bt[ar][ks8] = *(const uint4*)(Km + (size_t)(tile_n + ar) * ld + k0 + ks8);
        __syncthreads();

        v16bf af, bf0, bf1;
        {
            const __bf16* p = &As[wr * 16 + l15][0];
            af = mrg(*(const v8bf*)(p + khiA), *(const v8bf*)(p + 16 + khiA));
            const __bf16* q = &Bt[wc * 32 + l15][0];
            bf0 = mrg(*(const v8bf*)(q + khiA), *(const v8bf*)(q + 16 + khiA));
            q = &Bt[wc * 32 + 16 + l15][0];
            bf1 = mrg(*(const v8bf*)(q + khiA), *(const v8bf*)(q + 16 + khiA));
        }
        acc0 = __builtin_amdgcn_wmma_f32_16x16x32_bf16(false, af, false, bf0, (short)0, acc0, false, false);
        acc1 = __builtin_amdgcn_wmma_f32_16x16x32_bf16(false, af, false, bf1, (short)0, acc1, false, false);
        __syncthreads();
    }

    const int mb = tile_m + wr * 16 + ((lane >= 16) ? 8 : 0);
    const int n0 = tile_n + wc * 32 + l15;
    #pragma unroll
    for (int r = 0; r < 8; r++) {
        size_t ro = (size_t)(mb + r) * ldS;
        S[ro + n0]      = acc0[r] * scale;
        S[ro + n0 + 16] = acc1[r] * scale;
    }
}

// ---------------------------------------------------------------------------
// Host-side orchestration
// ---------------------------------------------------------------------------
extern "C" void kernel_launch(void* const* d_in, const int* in_sizes, int n_in,
                              void* d_out, int out_size, void* d_ws, size_t ws_size,
                              hipStream_t stream) {
    (void)in_sizes; (void)n_in; (void)out_size; (void)ws_size;

    const int*   x     = (const int*)  d_in[0];
    /* d_in[1] = lengths (unused by reference forward) */
    const float* emb   = (const float*)d_in[2];
    const float* pos   = (const float*)d_in[3];
    const float* ln1_g = (const float*)d_in[4];
    const float* ln1_b = (const float*)d_in[5];
    const float* ln2_g = (const float*)d_in[6];
    const float* ln2_b = (const float*)d_in[7];
    const float* Wq    = (const float*)d_in[8];
    const float* bq    = (const float*)d_in[9];
    const float* Wk    = (const float*)d_in[10];
    const float* bk    = (const float*)d_in[11];
    const float* Wv    = (const float*)d_in[12];
    const float* bv    = (const float*)d_in[13];
    const float* Wo    = (const float*)d_in[14];
    const float* bo    = (const float*)d_in[15];
    const float* W1    = (const float*)d_in[16];
    const float* b1    = (const float*)d_in[17];
    const float* W2    = (const float*)d_in[18];
    const float* b2    = (const float*)d_in[19];
    const float* lnf_g = (const float*)d_in[20];
    const float* lnf_b = (const float*)d_in[21];
    const float* Wdec  = (const float*)d_in[22];
    const float* bdec  = (const float*)d_in[23];

    const int T = 2048, D = 512, H = 8, DFF = 2048, V = 100, L = 6;
    const size_t TD = (size_t)T * D;
    const size_t DD = (size_t)D * D;

    // ---- workspace carve-up (256B aligned chunks) ----
    char* cur = (char*)d_ws;
    auto alloc = [&](size_t bytes) -> void* {
        void* r = (void*)cur;
        cur += (bytes + 255) & ~(size_t)255;
        return r;
    };
    float*  sc    = (float*) alloc((size_t)T * T * 4);  // fp32 scores (per head)
    __bf16* pb    = (__bf16*)alloc((size_t)T * T * 2);  // bf16 probs
    __bf16* hbuf  = (__bf16*)alloc(TD * 2);
    __bf16* hn    = (__bf16*)alloc(TD * 2);
    __bf16* qb    = (__bf16*)alloc(TD * 2);
    __bf16* kb    = (__bf16*)alloc(TD * 2);
    __bf16* vt    = (__bf16*)alloc(TD * 2);             // V transposed: [D][T]
    __bf16* yb    = (__bf16*)alloc(TD * 2);
    __bf16* mid   = (__bf16*)alloc((size_t)T * DFF * 2);
    __bf16* wqT   = (__bf16*)alloc(DD * 2);
    __bf16* wkT   = (__bf16*)alloc(DD * 2);
    __bf16* wvT   = (__bf16*)alloc(DD * 2);
    __bf16* woT   = (__bf16*)alloc(DD * 2);
    __bf16* w1T   = (__bf16*)alloc((size_t)D * DFF * 2);
    __bf16* w2T   = (__bf16*)alloc((size_t)DFF * D * 2);
    __bf16* wdecT = (__bf16*)alloc((size_t)V * D * 2);

    dim3 blk(NTHR);
    dim3 gD(T / 128, D / 64);     // (16, 8)
    dim3 gFF(T / 128, DFF / 64);  // (16, 32)
    dim3 gAV(T / 128, 1);
    dim3 gSC(T / 64, T / 64);     // (32, 32)
    dim3 gDec(T / 128, 2);        // V=100 -> 2 N-tiles (bounds-checked)
    dim3 cDD(D / 32, D / 32);     // (16,16) transpose 512x512
    dim3 cW1(DFF / 32, D / 32);   // (64,16) in[512][2048] -> out[2048][512]
    dim3 cW2(D / 32, DFF / 32);   // (16,64) in[2048][512] -> out[512][2048]
    dim3 cDec(4, D / 32);         // (4,16)  in[512][100]  -> out[100][512]
    const float scale = 0.125f;   // 1/sqrt(64)

    embed_kernel<<<T, blk, 0, stream>>>(x, emb, pos, hbuf);

    for (int l = 0; l < L; l++) {
        const float* bq_l = bq + (size_t)l * D;
        const float* bk_l = bk + (size_t)l * D;
        const float* bv_l = bv + (size_t)l * D;
        const float* bo_l = bo + (size_t)l * D;
        const float* b1_l = b1 + (size_t)l * DFF;
        const float* b2_l = b2 + (size_t)l * D;

        // ---- transpose+convert this layer's weights to bf16 ----
        convT_kernel<<<cDD, blk, 0, stream>>>(Wq + (size_t)l * DD, wqT, D, D);
        convT_kernel<<<cDD, blk, 0, stream>>>(Wk + (size_t)l * DD, wkT, D, D);
        convT_kernel<<<cDD, blk, 0, stream>>>(Wv + (size_t)l * DD, wvT, D, D);
        convT_kernel<<<cDD, blk, 0, stream>>>(Wo + (size_t)l * DD, woT, D, D);
        convT_kernel<<<cW1, blk, 0, stream>>>(W1 + (size_t)l * D * DFF, w1T, D, DFF);
        convT_kernel<<<cW2, blk, 0, stream>>>(W2 + (size_t)l * DFF * D, w2T, DFF, D);

        // hn = ln1(h)
        ln_kernel<<<T, blk, 0, stream>>>(hbuf, hn, ln1_g + (size_t)l * D, ln1_b + (size_t)l * D);

        // q, k row-major; v written TRANSPOSED into vt[D][T]
        gemm_wmma_kernel<<<gD, blk, 0, stream>>>(hn, D, wqT, bq_l, qb, D, nullptr, D, D, MODE_BF16, 0);
        gemm_wmma_kernel<<<gD, blk, 0, stream>>>(hn, D, wkT, bk_l, kb, D, nullptr, D, D, MODE_BF16, 0);
        gemm_wmma_kernel<<<gD, blk, 0, stream>>>(hn, D, wvT, bv_l, vt, T, nullptr, D, D, MODE_BF16T, 0);

        // attention, one head at a time
        for (int h = 0; h < H; h++) {
            attn_score_kernel<<<gSC, blk, 0, stream>>>(qb + h * 64, kb + h * 64, D, sc, T, scale);
            softmax_row_kernel<<<T, blk, 0, stream>>>(sc, pb, T);
            // y_h = probs @ V_h ; B operand = vt rows (already [dim][T])
            gemm_wmma_kernel<<<gAV, blk, 0, stream>>>(pb, T, vt + (size_t)h * 64 * T, nullptr,
                                                      yb + h * 64, D, nullptr, T, 64, MODE_BF16, 0);
        }

        // h = hn + y @ Wo + bo
        gemm_wmma_kernel<<<gD, blk, 0, stream>>>(yb, D, woT, bo_l, hbuf, D, hn, D, D, MODE_BF16, 0);

        // h2 = ln2(h); m = gelu(h2 @ W1 + b1); h = h + m @ W2 + b2
        ln_kernel<<<T, blk, 0, stream>>>(hbuf, hn, ln2_g + (size_t)l * D, ln2_b + (size_t)l * D);
        gemm_wmma_kernel<<<gFF, blk, 0, stream>>>(hn, D, w1T, b1_l, mid, DFF, nullptr, D, DFF, MODE_BF16, 1);
        gemm_wmma_kernel<<<gD, blk, 0, stream>>>(mid, DFF, w2T, b2_l, hbuf, D, hbuf, DFF, D, MODE_BF16, 0);
    }

    // final LN + decode (fp32 out)
    convT_kernel<<<cDec, blk, 0, stream>>>(Wdec, wdecT, D, V);
    ln_kernel<<<T, blk, 0, stream>>>(hbuf, hn, lnf_g, lnf_b);
    gemm_wmma_kernel<<<gDec, blk, 0, stream>>>(hn, D, wdecT, bdec, d_out, V,
                                               nullptr, D, V, MODE_F32, 0);
}
